// ShardedBTLMAttention_70102456205743
// MI455X (gfx1250) — matrix operand order repaired and checked by reading, not compiled, not run on hardware
//
#include <hip/hip_runtime.h>
#include <hip/hip_bf16.h>

typedef __attribute__((ext_vector_type(16))) __bf16 v16bf;
typedef __attribute__((ext_vector_type(8)))  __bf16 v8bf;
typedef __attribute__((ext_vector_type(4)))  __bf16 v4bf;
typedef __attribute__((ext_vector_type(8)))  float  v8f;

// gcc-style int4 vector matching the async-LDS builtin's parameter type:
//   void (v4i addrspace(1)*, v4i addrspace(3)*, imm int, imm int)
typedef int v4i_t __attribute__((vector_size(16)));
typedef __attribute__((address_space(1))) v4i_t g_v4i;
typedef __attribute__((address_space(3))) v4i_t l_v4i;

#if __has_builtin(__builtin_amdgcn_global_load_async_to_lds_b128) && \
    __has_builtin(__builtin_amdgcn_s_wait_asynccnt)
#define HAVE_ASYNC_LDS 1
#endif

#if __has_builtin(__builtin_amdgcn_permlane16)
#define HAVE_PERMLANE16 1
#endif

__device__ inline __bf16 f2bf(float f) {
  unsigned u = __builtin_bit_cast(unsigned, f);
  unsigned r = u + 0x7FFFu + ((u >> 16) & 1u);
  unsigned short h = (unsigned short)(r >> 16);
  return __builtin_bit_cast(__bf16, h);
}

__device__ inline __bf16 bf_zero() {
  return __builtin_bit_cast(__bf16, (unsigned short)0);
}

// Build a v16bf A/B fragment from two contiguous 8-element halves in LDS.
__device__ inline v16bf ld16(const __bf16* p0, const __bf16* p1) {
  v8bf lo = *(const v8bf*)p0;
  v8bf hi = *(const v8bf*)p1;
  return __builtin_shufflevector(lo, hi, 0,1,2,3,4,5,6,7,8,9,10,11,12,13,14,15);
}

// ---------------------------------------------------------------------------
// 16-lane-row butterfly reductions. Prefer V_PERMLANE16_B32 (VALU) over
// __shfl_xor (ds_bpermute: LDS pipe, which also feeds WMMA fragments).
// Lane-select nibble i of {S2,S1} = i ^ step.
// ---------------------------------------------------------------------------
#ifdef HAVE_PERMLANE16
__device__ inline float pl16(float x, unsigned s1, unsigned s2) {
  int xi = __builtin_bit_cast(int, x);
  int r  = __builtin_amdgcn_permlane16(xi, xi, s1, s2, false, false);
  return __builtin_bit_cast(float, r);
}
#endif

__device__ inline float row_reduce_max(float x) {
#ifdef HAVE_PERMLANE16
  x = fmaxf(x, pl16(x, 0x67452301u, 0xEFCDAB89u));  // xor 1
  x = fmaxf(x, pl16(x, 0x54761032u, 0xDCFE98BAu));  // xor 2
  x = fmaxf(x, pl16(x, 0x32107654u, 0xBA98FEDCu));  // xor 4
  x = fmaxf(x, pl16(x, 0xFEDCBA98u, 0x76543210u));  // xor 8
#else
#pragma unroll
  for (int off = 8; off >= 1; off >>= 1) x = fmaxf(x, __shfl_xor(x, off, 32));
#endif
  return x;
}

__device__ inline float row_reduce_sum(float x) {
#ifdef HAVE_PERMLANE16
  x += pl16(x, 0x67452301u, 0xEFCDAB89u);  // xor 1
  x += pl16(x, 0x54761032u, 0xDCFE98BAu);  // xor 2
  x += pl16(x, 0x32107654u, 0xBA98FEDCu);  // xor 4
  x += pl16(x, 0xFEDCBA98u, 0x76543210u);  // xor 8
#else
#pragma unroll
  for (int off = 8; off >= 1; off >>= 1) x += __shfl_xor(x, off, 32);
#endif
  return x;
}

// ---------------------------------------------------------------------------
// Prep kernel 1: elementwise f32 -> bf16 (hidden_states)
// ---------------------------------------------------------------------------
__global__ void __launch_bounds__(256)
convert_f32_bf16(const float* __restrict__ src, __bf16* __restrict__ dst, int n) {
  int i = (blockIdx.x * 256 + threadIdx.x) * 4;
  if (i < n) {
    float4 f = *(const float4*)(&src[i]);
    v4bf o;
    o[0] = f2bf(f.x); o[1] = f2bf(f.y); o[2] = f2bf(f.z); o[3] = f2bf(f.w);
    *(v4bf*)(&dst[i]) = o;
  }
}

// ---------------------------------------------------------------------------
// Prep kernel 2: W[K][N] f32 -> Wt[N][K] bf16 (64x64 LDS tiles, coalesced
// on both sides). Done once so GEMM hot loops have zero conversion VALU.
// ---------------------------------------------------------------------------
#define TT 64
__global__ void __launch_bounds__(256)
transpose_convert(const float* __restrict__ W, __bf16* __restrict__ Wt,
                  int K, int N) {
  __shared__ __bf16 T[TT][TT + 8];
  const int tid = threadIdx.x;
  const int k0 = blockIdx.y * TT;
  const int n0 = blockIdx.x * TT;
  for (int i = tid; i < TT * TT; i += 256) {
    int r = i >> 6, c = i & 63;
    T[c][r] = f2bf(W[(size_t)(k0 + r) * N + n0 + c]);
  }
  __syncthreads();
  for (int i = tid; i < TT * TT; i += 256) {
    int r = i >> 6, c = i & 63;
    Wt[(size_t)(n0 + r) * K + k0 + c] = T[r][c];
  }
}

// ---------------------------------------------------------------------------
// All-bf16 GEMM: C[M,N] = A[M,K] @ Bt[N,K]^T + bias[N]
// A row-major bf16, Bt K-major bf16 (pre-transposed weights).
// 8 waves (4M x 2N), wave tile 32x64, double-buffered async LDS staging.
// ---------------------------------------------------------------------------
#define GBM 128
#define GBN 128
#define GBK 32

template <typename TC>
__global__ void __launch_bounds__(256)
gemm_bf16(const __bf16* __restrict__ A, const __bf16* __restrict__ Bt,
          const float* __restrict__ bias, TC* __restrict__ C,
          int M, int N, int K) {
  __shared__ __align__(16) __bf16 As[2][GBM][GBK];   // 2 x 8KB
  __shared__ __align__(16) __bf16 Bs[2][GBN][GBK];   // 2 x 8KB

  const int tid  = threadIdx.x;
  const int lane = tid & 31;
  const int w    = tid >> 5;
  const int wm   = w >> 1;          // 0..3 -> M offset (x32)
  const int wn   = w & 1;           // 0..1 -> N offset (x64)
  const int l16  = lane & 15;
  const int lh   = lane >> 4;
  const int bm   = blockIdx.y * GBM;
  const int bn   = blockIdx.x * GBN;
  const int nkt  = K / GBK;

  // Stage tile kt into LDS buffer bsel: 2 x 16B per thread per array.
  auto stage = [&](int kt, int bsel) {
#pragma unroll
    for (int i = 0; i < 2; ++i) {
      int li  = tid + 256 * i;
      int row = li >> 2;
      int c8  = (li & 3) * 8;
      const __bf16* ga = &A [(size_t)(bm + row) * K + kt * GBK + c8];
      const __bf16* gb = &Bt[(size_t)(bn + row) * K + kt * GBK + c8];
#ifdef HAVE_ASYNC_LDS
      __builtin_amdgcn_global_load_async_to_lds_b128(
          (g_v4i*)ga, (l_v4i*)&As[bsel][row][c8], 0, 0);
      __builtin_amdgcn_global_load_async_to_lds_b128(
          (g_v4i*)gb, (l_v4i*)&Bs[bsel][row][c8], 0, 0);
#else
      *(v8bf*)&As[bsel][row][c8] = *(const v8bf*)ga;
      *(v8bf*)&Bs[bsel][row][c8] = *(const v8bf*)gb;
#endif
    }
  };

  v8f z = {};
  v8f acc[2][4];
#pragma unroll
  for (int i = 0; i < 2; ++i)
#pragma unroll
    for (int j = 0; j < 4; ++j) acc[i][j] = z;

  stage(0, 0);
#ifdef HAVE_ASYNC_LDS
  __builtin_amdgcn_s_wait_asynccnt(0);
#endif
  __syncthreads();

  for (int kt = 0; kt < nkt; ++kt) {
    const int b = kt & 1;
    // Prefetch next tile into the other buffer while we compute this one.
    if (kt + 1 < nkt) stage(kt + 1, (kt + 1) & 1);

    v16bf a[2], bb[4];
#pragma unroll
    for (int mt = 0; mt < 2; ++mt) {
      const int row = wm * 32 + mt * 16 + l16;
      a[mt] = ld16(&As[b][row][lh * 8], &As[b][row][lh * 8 + 16]);
    }
#pragma unroll
    for (int nt = 0; nt < 4; ++nt) {
      const int col = wn * 64 + nt * 16 + l16;
      bb[nt] = ld16(&Bs[b][col][lh * 8], &Bs[b][col][lh * 8 + 16]);
    }
#pragma unroll
    for (int mt = 0; mt < 2; ++mt)
#pragma unroll
      for (int nt = 0; nt < 4; ++nt)
        acc[mt][nt] = __builtin_amdgcn_wmma_f32_16x16x32_bf16(
            false, a[mt], false, bb[nt], (short)0, acc[mt][nt], false, false);

#ifdef HAVE_ASYNC_LDS
    __builtin_amdgcn_s_wait_asynccnt(0);
#endif
    __syncthreads();
  }

  // Epilogue: D layout -> row m = r + 8*(lane/16), col n = lane%16
#pragma unroll
  for (int mt = 0; mt < 2; ++mt)
#pragma unroll
    for (int nt = 0; nt < 4; ++nt)
#pragma unroll
      for (int r = 0; r < 8; ++r) {
        int row = bm + wm * 32 + mt * 16 + r + lh * 8;
        int col = bn + wn * 64 + nt * 16 + l16;
        float v = acc[mt][nt][r] + bias[col];
        if constexpr (sizeof(TC) == 4) C[(size_t)row * N + col] = v;
        else                           C[(size_t)row * N + col] = f2bf(v);
      }
}

// ---------------------------------------------------------------------------
// Flash attention over bf16 QKV: grid (S/64, heads), 4 waves, 16 q-rows/wave.
// ---------------------------------------------------------------------------
#define S_LEN 2048
#define EMB   2560
#define QKV_LD 7680
#define HDIM  80
#define HPAD  96

__global__ void __launch_bounds__(128)
flash_attn(const __bf16* __restrict__ qkv, const float* __restrict__ pos_bias,
           const float* __restrict__ amask, __bf16* __restrict__ out) {
  __shared__ __align__(16) __bf16 Qs[4][16][HPAD];  // per-wave Q, zero-padded d
  __shared__ __align__(16) __bf16 Ks[32][HPAD];     // K tile row-major, padded
  __shared__ __align__(16) __bf16 Vt[HDIM][32];     // V tile transposed (d, k)
  __shared__ __align__(16) __bf16 Ps[4][16][32];    // per-wave P tile

  const int tid  = threadIdx.x;
  const int lane = tid & 31;
  const int w    = tid >> 5;
  const int l16  = lane & 15;
  const int lh   = lane >> 4;
  const int h    = blockIdx.y;
  const int qbase = blockIdx.x * 64 + w * 16;
  const __bf16 zb = bf_zero();

  // Zero the pad columns of Ks once (never overwritten by staging).
  for (int i = tid; i < 32 * 16; i += 128) {
    Ks[i >> 4][80 + (i & 15)] = zb;
  }

  // Stage this wave's 16x96 Q tile: 10 x 16B copies per row + zero pad.
  for (int i = lane; i < 16 * 10; i += 32) {
    int row = i / 10, c = i % 10;
    *(v8bf*)&Qs[w][row][c * 8] =
        *(const v8bf*)&qkv[(size_t)(qbase + row) * QKV_LD + h * HDIM + c * 8];
  }
  for (int i = lane; i < 16 * 16; i += 32) {
    Qs[w][i >> 4][80 + (i & 15)] = zb;
  }
  v16bf qf[3];
#pragma unroll
  for (int kd = 0; kd < 3; ++kd) {
    int kb = kd * 32 + lh * 8;
    qf[kd] = ld16(&Qs[w][l16][kb], &Qs[w][l16][kb + 16]);
  }

  v8f z = {};
  v8f O[5];
  float run_max[8], run_sum[8];
#pragma unroll
  for (int j = 0; j < 5; ++j) O[j] = z;
#pragma unroll
  for (int r = 0; r < 8; ++r) { run_max[r] = -3.0e38f; run_sum[r] = 0.f; }

  const size_t bias_base = (size_t)h * S_LEN * S_LEN;

  for (int kt = 0; kt < S_LEN / 32; ++kt) {
    const int kbase = kt * 32;
    __syncthreads();

    // ---- Stage K tile (32 rows x 80 bf16 = 10 x 16B chunks per row) ----
#ifdef HAVE_ASYNC_LDS
    for (int i = tid; i < 320; i += 128) {
      int r = i / 10, c = i % 10;
      const __bf16* gp =
          &qkv[(size_t)(kbase + r) * QKV_LD + EMB + h * HDIM + c * 8];
      __builtin_amdgcn_global_load_async_to_lds_b128(
          (g_v4i*)gp, (l_v4i*)&Ks[r][c * 8], 0, 0);
    }
#else
    for (int i = tid; i < 320; i += 128) {
      int r = i / 10, c = i % 10;
      *(v8bf*)&Ks[r][c * 8] =
          *(const v8bf*)&qkv[(size_t)(kbase + r) * QKV_LD + EMB + h * HDIM + c * 8];
    }
#endif
    // ---- Stage V tile transposed: coalesced 16B reads, LDS scatter ----
    for (int i = tid; i < 320; i += 128) {
      int r = i / 10, c = i % 10;
      v8bf v = *(const v8bf*)&qkv[(size_t)(kbase + r) * QKV_LD + 2 * EMB +
                                  h * HDIM + c * 8];
#pragma unroll
      for (int j = 0; j < 8; ++j) Vt[c * 8 + j][r] = v[j];
    }
    // Prefetch next k-tile's position-bias rows (dominant HBM stream).
    if (kt + 1 < S_LEN / 32) {
      const size_t nb = bias_base + (size_t)(qbase + lh * 8) * S_LEN + kbase + 32;
      __builtin_prefetch(&pos_bias[nb + (size_t)l16 * 8], 0, 1);
    }
#ifdef HAVE_ASYNC_LDS
    __builtin_amdgcn_s_wait_asynccnt(0);
#endif
    __syncthreads();

    // ---- scores = Q @ K^T over padded depth 96 ----
    v8f s0 = z, s1 = z;
#pragma unroll
    for (int kd = 0; kd < 3; ++kd) {
      int kb = kd * 32 + lh * 8;
      v16bf b0 = ld16(&Ks[l16][kb], &Ks[l16][kb + 16]);
      v16bf b1 = ld16(&Ks[16 + l16][kb], &Ks[16 + l16][kb + 16]);
      s0 = __builtin_amdgcn_wmma_f32_16x16x32_bf16(false, qf[kd], false, b0,
                                                   (short)0, s0, false, false);
      s1 = __builtin_amdgcn_wmma_f32_16x16x32_bf16(false, qf[kd], false, b1,
                                                   (short)0, s1, false, false);
    }

    // ---- Online softmax in D layout (row lives in a 16-lane group) ----
#pragma unroll
    for (int r = 0; r < 8; ++r) {
      const int m = r + lh * 8;
      const size_t rowb = bias_base + (size_t)(qbase + m) * S_LEN + kbase;
      const int c0 = l16, c1 = 16 + l16;
      float t0 = s0[r] * (1.f / 80.f) + pos_bias[rowb + c0] + amask[kbase + c0];
      float t1 = s1[r] * (1.f / 80.f) + pos_bias[rowb + c1] + amask[kbase + c1];
      float mx = row_reduce_max(fmaxf(t0, t1));
      float mnew  = fmaxf(run_max[r], mx);
      float alpha = __expf(run_max[r] - mnew);
      float p0 = __expf(t0 - mnew);
      float p1 = __expf(t1 - mnew);
      float ls = row_reduce_sum(p0 + p1);
      run_sum[r] = run_sum[r] * alpha + ls;
      run_max[r] = mnew;
#pragma unroll
      for (int j = 0; j < 5; ++j) O[j][r] *= alpha;
      Ps[w][m][c0] = f2bf(p0);
      Ps[w][m][c1] = f2bf(p1);
    }

    // ---- O += P @ V ----
    {
      int kb = lh * 8;
      v16bf pa = ld16(&Ps[w][l16][kb], &Ps[w][l16][kb + 16]);
#pragma unroll
      for (int j = 0; j < 5; ++j) {
        v16bf bv = ld16(&Vt[j * 16 + l16][lh * 8], &Vt[j * 16 + l16][lh * 8 + 16]);
        O[j] = __builtin_amdgcn_wmma_f32_16x16x32_bf16(false, pa, false, bv,
                                                       (short)0, O[j], false, false);
      }
    }
  }

  // Normalize and write merged-head output [S, E] as bf16.
#pragma unroll
  for (int r = 0; r < 8; ++r) {
    const int m = r + lh * 8;
    const float inv = 1.f / run_sum[r];
#pragma unroll
    for (int j = 0; j < 5; ++j)
      out[(size_t)(qbase + m) * EMB + h * HDIM + j * 16 + l16] =
          f2bf(O[j][r] * inv);
  }
}

extern "C" void kernel_launch(void* const* d_in, const int* in_sizes, int n_in,
                              void* d_out, int out_size, void* d_ws, size_t ws_size,
                              hipStream_t stream) {
  const float* hs   = (const float*)d_in[0];  // [1,2048,2560]
  const float* mask = (const float*)d_in[1];  // [1,1,1,2048]
  const float* pb   = (const float*)d_in[2];  // [32,2048,2048]
  const float* w1   = (const float*)d_in[3];  // [2560,7680]
  const float* b1   = (const float*)d_in[4];  // [7680]
  const float* w2   = (const float*)d_in[5];  // [2560,2560]
  const float* b2   = (const float*)d_in[6];  // [2560]
  float* out = (float*)d_out;

  // Workspace layout (bf16 elements), ~105 MB total:
  __bf16* X    = (__bf16*)d_ws;               // 2048*2560
  __bf16* W1t  = X    + (size_t)2048 * 2560;  // 7680*2560 (K-major)
  __bf16* W2t  = W1t  + (size_t)7680 * 2560;  // 2560*2560 (K-major)
  __bf16* qkv  = W2t  + (size_t)2560 * 2560;  // 2048*7680
  __bf16* attn = qkv  + (size_t)2048 * 7680;  // 2048*2560

  // Prep: one-time conversions (bandwidth-bound)
  convert_f32_bf16<<<(2048 * 2560) / 1024, 256, 0, stream>>>(hs, X, 2048 * 2560);
  transpose_convert<<<dim3(7680 / TT, 2560 / TT), 256, 0, stream>>>(w1, W1t, 2560, 7680);
  transpose_convert<<<dim3(2560 / TT, 2560 / TT), 256, 0, stream>>>(w2, W2t, 2560, 2560);

  // 1) QKV projection: all-bf16 GEMM, bf16 out
  gemm_bf16<__bf16><<<dim3(7680 / GBN, 2048 / GBM), 256, 0, stream>>>(
      X, W1t, b1, qkv, 2048, 7680, 2560);

  // 2) Flash attention (ALiBi bias + additive mask, 1/D scaling), bf16 I/O
  flash_attn<<<dim3(2048 / 64, 32), 128, 0, stream>>>(qkv, pb, mask, attn);

  // 3) Output projection: all-bf16 GEMM, f32 out
  gemm_bf16<float><<<dim3(2560 / GBN, 2048 / GBM), 256, 0, stream>>>(
      attn, W2t, b2, out, 2048, 2560, 2560);
}
